// GraphAttention_43576738185991
// MI455X (gfx1250) — matrix operand order repaired and checked by reading, not compiled, not run on hardware
//
#include <hip/hip_runtime.h>
#include <hip/hip_bf16.h>

// ---------------------------------------------------------------------------
// GAT layer on MI455X (gfx1250):
//   K1: feats = X @ W[h]              (WMMA bf16, fp32 accum)
//   K2: a_s/a_n = feats @ attn vecs
//   K3: fused flash-softmax(leakyrelu(a_s+a_n^T)+mask) @ feats  (WMMA bf16)
//       with double-buffered GLOBAL_LOAD_ASYNC_TO_LDS staging of the
//       adjacency stream (the only HBM-bound traffic: 67 MB @ 23.3 TB/s).
// ---------------------------------------------------------------------------

typedef __attribute__((ext_vector_type(16))) __bf16 v16bf;
typedef __attribute__((ext_vector_type(8)))  __bf16 v8bf;
typedef __attribute__((ext_vector_type(8)))  float  v8f;
typedef __attribute__((ext_vector_type(4)))  float  v4f;

#define GAT_N    4096
#define GAT_DIN  512
#define GAT_DH   128
#define GAT_H    4
#define LEAKY    0.2f
#define NEGBIG   1.0e10f

// Raw LDS byte offset of a __shared__ object (addrspace(3) pointers are
// 32-bit on amdgcn; generic->LDS addrspacecast strips the aperture).
typedef const void __attribute__((address_space(3))) lds_cvoid;
__device__ __forceinline__ unsigned lds_off(const void* p) {
  return (unsigned)(unsigned long long)(lds_cvoid*)p;
}

// ---------------------------------------------------------------------------
// Kernel 1: per-head projection. Grid: (H * N/16) blocks x 256 threads.
// Each wave (8 per block) owns one 16x16 output tile (wave id = col tile).
// A (16x32) and B (32x16) loaded in the same contiguous k order:
//   A: lane m=l%16, elem  e    -> k = kh*16+e
//   B: lane n=l%16, elem  e    -> k = kh*16+e          (kh = l>>4)
// Outputs: featsF fp32 [h][node][c]  and  featsBT bf16 [h][c][node].
// ---------------------------------------------------------------------------
__global__ __launch_bounds__(256)
void gat_proj(const float* __restrict__ X, const float* __restrict__ W,
              float* __restrict__ featsF, __bf16* __restrict__ featsBT) {
  const int b       = blockIdx.x;
  const int h       = b >> 8;             // 256 row tiles per head
  const int rowbase = (b & 255) << 4;
  const int wave    = threadIdx.x >> 5;
  const int lane    = threadIdx.x & 31;
  const int m       = lane & 15;
  const int hi      = lane >> 4;
  const int ncol    = wave * 16 + m;      // feature column of this lane

  const float* Wh   = W + (size_t)h * GAT_DIN * GAT_DH;
  const float* xrow = X + (size_t)(rowbase + m) * GAT_DIN;

  v8f acc = {};
  for (int ks = 0; ks < GAT_DIN; ks += 32) {
    v16bf a;
    const float* ap0 = xrow + ks + hi * 16;       // k = ks + hi*16 + e
    const float* ap1 = xrow + ks + hi * 16 + 8;   // k = ks + hi*16 + 8 + e
#pragma unroll
    for (int e = 0; e < 8; ++e) {
      a[e]     = (__bf16)ap0[e];
      a[8 + e] = (__bf16)ap1[e];
    }
    v16bf bm;
    const float* bp = Wh + (size_t)(ks + hi * 16) * GAT_DH + ncol;
#pragma unroll
    for (int e = 0; e < 16; ++e) bm[e] = (__bf16)bp[(size_t)e * GAT_DH];

    acc = __builtin_amdgcn_wmma_f32_16x16x32_bf16(false, a, false, bm,
                                                  (short)0, acc, false, false);
  }

  // D layout: lane holds col n = ncol, rows r + 8*hi for r = 0..7
  float* fO = featsF + ((size_t)h * GAT_N + rowbase) * GAT_DH + ncol;
  v8bf pk;
#pragma unroll
  for (int r = 0; r < 8; ++r) {
    const int mr = r + hi * 8;
    fO[(size_t)mr * GAT_DH] = acc[r];
    pk[r] = (__bf16)acc[r];
  }
  // transposed bf16: rows are contiguous -> single 16B store
  __bf16* bt = featsBT + ((size_t)h * GAT_DH + ncol) * GAT_N + rowbase + hi * 8;
  *(v8bf*)bt = pk;
}

// ---------------------------------------------------------------------------
// Kernel 2: attention logits a_s[h][n], a_n[h][n] (128-dim dots).
// ---------------------------------------------------------------------------
__global__ __launch_bounds__(256)
void gat_logits(const float* __restrict__ featsF,
                const float* __restrict__ attn_s, const float* __restrict__ attn_n,
                float* __restrict__ a_s, float* __restrict__ a_n) {
  const int t = blockIdx.x * blockDim.x + threadIdx.x;   // 0 .. H*N-1
  const int h = t >> 12;
  const float* f  = featsF + (size_t)t * GAT_DH;
  const float* as = attn_s + h * GAT_DH;
  const float* an = attn_n + h * GAT_DH;
  float s0 = 0.f, s1 = 0.f;
#pragma unroll 8
  for (int c = 0; c < GAT_DH; c += 4) {
    const v4f fv = *(const v4f*)(f + c);
    const v4f av = *(const v4f*)(as + c);
    const v4f nv = *(const v4f*)(an + c);
    s0 += fv.x * av.x + fv.y * av.y + fv.z * av.z + fv.w * av.w;
    s1 += fv.x * nv.x + fv.y * nv.y + fv.z * nv.z + fv.w * nv.w;
  }
  a_s[t] = s0;
  a_n[t] = s1;
}

// ---------------------------------------------------------------------------
// Async DMA of one j-tile into LDS (issued by wave 0 only, all 32 lanes):
//   adjacency 16x32 fp32 (2 KB) = 4 x b128/lane issues, non-temporal
//   a_n slice  4x32  fp32 (512B) = 1 x b128/lane issue
// Tracked by ASYNCcnt; completion enforced with s_wait_asynccnt + barrier.
// ---------------------------------------------------------------------------
__device__ __forceinline__ void stage_tile_async(
    unsigned ldsAdj, unsigned ldsAn,
    const float* __restrict__ adjRowBase,   // adj + rowbase*N
    const float* __restrict__ a_n,
    int jb, int lane) {
  const int rowA = lane >> 3;         // + 4*i
  const int colA = (lane & 7) << 2;   // 4-float chunk
#pragma unroll
  for (int i = 0; i < 4; ++i) {
    const unsigned dst = ldsAdj + (unsigned)((((i * 4 + rowA) * 32) + colA) * 4);
    const int voff = ((i * 4 + rowA) * GAT_N + jb + colA) * 4;
    asm volatile("global_load_async_to_lds_b128 %0, %1, %2 th:TH_LOAD_NT"
                 :: "v"(dst), "v"(voff), "s"(adjRowBase) : "memory");
  }
  const unsigned dstN  = ldsAn + (unsigned)(lane * 16);  // [head][32] layout
  const int      voffN = ((lane >> 3) * GAT_N + jb + colA) * 4;
  asm volatile("global_load_async_to_lds_b128 %0, %1, %2"
               :: "v"(dstN), "v"(voffN), "s"(a_n) : "memory");
}

// ---------------------------------------------------------------------------
// Kernel 3: fused flash-softmax + aggregation.
// Grid: 256 blocks (16-row tiles) x 256 threads (8 waves).
// Wave w -> head w/2, feature half (w&1)*64 (4 WMMA col tiles).
// All 4 heads share one adjacency tile per block -> adjacency streamed from
// HBM exactly once. S is built directly in WMMA A-layout; P=exp(S-m) in bf16
// feeds v_wmma_f32_16x16x32_bf16 against contiguous feature-major V tiles
// (featsBT is 4 MB total -> L2-resident on the 192 MB L2).
// ---------------------------------------------------------------------------
__global__ __launch_bounds__(256)
void gat_attn(const float* __restrict__ adj, const float* __restrict__ a_s,
              const float* __restrict__ a_n, const __bf16* __restrict__ featsBT,
              const float* __restrict__ bias, float* __restrict__ out) {
  __shared__ float adjT[2][16 * 32];
  __shared__ float anT[2][GAT_H * 32];

  const int rowbase = blockIdx.x << 4;
  const int tid     = threadIdx.x;
  const int wave    = tid >> 5;
  const int lane    = tid & 31;
  const int h       = wave >> 1;
  const int colbase = (wave & 1) * 64;
  const int m       = lane & 15;
  const int hi      = lane >> 4;

  const float* adjRowBase = adj + (size_t)rowbase * GAT_N;
  const unsigned ldsAdj0 = lds_off(&adjT[0][0]);
  const unsigned ldsAdj1 = lds_off(&adjT[1][0]);
  const unsigned ldsAn0  = lds_off(&anT[0][0]);
  const unsigned ldsAn1  = lds_off(&anT[1][0]);

  const float asr = a_s[h * GAT_N + rowbase + m];   // this lane's row logit

  float mrun = -3.0e38f;
  float lrun = 0.0f;
  v8f acc[4] = {{}, {}, {}, {}};

  int buf = 0;
  if (wave == 0)
    stage_tile_async(ldsAdj0, ldsAn0, adjRowBase, a_n, 0, lane);

#pragma unroll 1
  for (int jb = 0; jb < GAT_N; jb += 32) {
    if (wave == 0)
      asm volatile("s_wait_asynccnt 0x0" ::: "memory");   // current tile landed
    __syncthreads();
    if (wave == 0 && jb + 32 < GAT_N)                     // prefetch next tile
      stage_tile_async(buf ? ldsAdj0 : ldsAdj1, buf ? ldsAn0 : ldsAn1,
                       adjRowBase, a_n, jb + 32, lane);

    // --- build S tile in WMMA A-layout (16 values / lane) ---
    float s[16];
    const float* arow = adjT[buf] + m * 32 + hi * 16;
    const float* anh  = anT[buf] + h * 32 + hi * 16;
#pragma unroll
    for (int e = 0; e < 8; ++e) {
      const float ad0 = arow[e], ad1 = arow[8 + e];
      const float an0 = anh[e],  an1 = anh[8 + e];
      float t0 = asr + an0; t0 = (t0 > 0.f) ? t0 : LEAKY * t0;
      float t1 = asr + an1; t1 = (t1 > 0.f) ? t1 : LEAKY * t1;
      s[e]     = fmaf(ad0 - 1.0f, NEGBIG, t0);   // mask: -1e10 where adj==0
      s[8 + e] = fmaf(ad1 - 1.0f, NEGBIG, t1);
    }

    // --- online softmax: row max over lane + partner (lane ^ 16) ---
    float lm = s[0];
#pragma unroll
    for (int e = 1; e < 16; ++e) lm = fmaxf(lm, s[e]);
    lm = fmaxf(lm, __shfl_xor(lm, 16));
    const float mnew  = fmaxf(mrun, lm);
    const float scale = __expf(mrun - mnew);
    mrun = mnew;

    v16bf pa;
    float ls = 0.f;
#pragma unroll
    for (int e = 0; e < 16; ++e) {
      const float p = __expf(s[e] - mnew);
      ls += p;
      pa[e] = (__bf16)p;
    }
    ls += __shfl_xor(ls, 16);
    lrun = lrun * scale + ls;

    // --- rescale accumulators (C-layout rows r + 8*hi) ---
#pragma unroll
    for (int r = 0; r < 8; ++r) {
      const float sc = __shfl(scale, r + hi * 8);
      acc[0][r] *= sc; acc[1][r] *= sc; acc[2][r] *= sc; acc[3][r] *= sc;
    }

    // --- acc += P (16x32) x V (32x16) per feature tile ---
#pragma unroll
    for (int t4 = 0; t4 < 4; ++t4) {
      const int c = colbase + t4 * 16 + m;
      const v16bf bv = *(const v16bf*)(featsBT + ((size_t)h * GAT_DH + c) * GAT_N
                                       + jb + hi * 16);
      acc[t4] = __builtin_amdgcn_wmma_f32_16x16x32_bf16(false, pa, false, bv,
                                                        (short)0, acc[t4],
                                                        false, false);
    }
    buf ^= 1;
  }

  // --- normalize + bias + store: out[node][h*128 + c] ---
#pragma unroll
  for (int r = 0; r < 8; ++r) {
    const float denom = __shfl(lrun, r + hi * 8);
    const float inv   = 1.0f / denom;
    const int node    = rowbase + r + hi * 8;
#pragma unroll
    for (int t4 = 0; t4 < 4; ++t4) {
      const int c = colbase + t4 * 16 + m;
      out[(size_t)node * (GAT_H * GAT_DH) + h * GAT_DH + c] =
          acc[t4][r] * inv + bias[h * GAT_DH + c];
    }
  }
}

// ---------------------------------------------------------------------------
extern "C" void kernel_launch(void* const* d_in, const int* in_sizes, int n_in,
                              void* d_out, int out_size, void* d_ws, size_t ws_size,
                              hipStream_t stream) {
  const float* adj    = (const float*)d_in[0];   // [N,N] fp32 0/1
  const float* X      = (const float*)d_in[1];   // [N,512]
  const float* W      = (const float*)d_in[2];   // [H,512,128]
  const float* bias   = (const float*)d_in[3];   // [H,128]
  const float* attn_s = (const float*)d_in[4];   // [H,128]
  const float* attn_n = (const float*)d_in[5];   // [H,128]
  float* out = (float*)d_out;                    // [N, H*128]

  char* ws = (char*)d_ws;
  const size_t featsElems = (size_t)GAT_H * GAT_N * GAT_DH;   // 2,097,152
  float*  featsF  = (float*)ws;                               // 8 MiB
  __bf16* featsBT = (__bf16*)(ws + featsElems * sizeof(float));        // 4 MiB
  float*  a_s     = (float*)(ws + featsElems * (sizeof(float) + sizeof(__bf16)));
  float*  a_n     = a_s + (size_t)GAT_H * GAT_N;

  gat_proj  <<<dim3(GAT_H * (GAT_N / 16)), dim3(256), 0, stream>>>(X, W, featsF, featsBT);
  gat_logits<<<dim3((GAT_H * GAT_N) / 256), dim3(256), 0, stream>>>(featsF, attn_s, attn_n, a_s, a_n);
  gat_attn  <<<dim3(GAT_N / 16), dim3(256), 0, stream>>>(adj, a_s, a_n, featsBT, bias, out);
}
